// PyTorchNativeAttention_1855425872537
// MI455X (gfx1250) — matrix operand order repaired and checked by reading, not compiled, not run on hardware
//
#include <hip/hip_runtime.h>

// ---- problem constants (match reference setup_inputs) ----
#define B_    2
#define S_    2048
#define H_    32
#define KVH_  8
#define D_    128
#define GQA_  (H_ / KVH_)
#define SCALE_ 0.08838834764831845f   // 1/sqrt(128)

typedef __attribute__((ext_vector_type(16))) __bf16 v16bf;
typedef __attribute__((ext_vector_type(8)))  float  v8f;

union Frag16 {
  v16bf v;
  uint4 q[2];
};

// LDS row strides (in bf16 elements); byte strides are multiples of 16 for b128 loads
#define LKS 136   // K tile: 128 + 8 pad  -> 272 B/row
#define LVS 40    // V^T tile: 32 + 8 pad ->  80 B/row

static __device__ __forceinline__ unsigned pk2(float a, float b) {
  union { __bf16 h[2]; unsigned u; } c;
  c.h[0] = (__bf16)a;
  c.h[1] = (__bf16)b;
  return c.u;
}

// Online-softmax update for one 16x32 score tile pair; MASKED specialization is
// only taken for the 1-2 k-blocks per wave that straddle the causal diagonal,
// so the hot path carries no v_cmp/v_cndmask work. needMask is wave-uniform ->
// scalar branch, EXEC stays full everywhere around the WMMAs.
template <bool MASKED>
static __device__ __forceinline__ void softmax_update(
    const v8f& s0, const v8f& s1, float* m_i, float* l_i, float* corr,
    __bf16* lPw, int nlo, int hhi, int qw, int keybase) {
#pragma unroll
  for (int r = 0; r < 8; ++r) {
    float a = s0[r], bb = s1[r];
    if (MASKED) {
      const int qrow_i = qw + r + 8 * hhi;
      if (keybase + nlo      > qrow_i) a  = -1e30f;
      if (keybase + 16 + nlo > qrow_i) bb = -1e30f;
    }
    float rm = fmaxf(a, bb);
    rm = fmaxf(rm, __shfl_xor(rm, 1, 32));
    rm = fmaxf(rm, __shfl_xor(rm, 2, 32));
    rm = fmaxf(rm, __shfl_xor(rm, 4, 32));
    rm = fmaxf(rm, __shfl_xor(rm, 8, 32));
    const float mn = fmaxf(m_i[r], rm);
    const float c  = __expf(m_i[r] - mn);
    m_i[r] = mn;
    const float pa = __expf(a - mn);
    const float pb = __expf(bb - mn);
    float rs = pa + pb;
    rs += __shfl_xor(rs, 1, 32);
    rs += __shfl_xor(rs, 2, 32);
    rs += __shfl_xor(rs, 4, 32);
    rs += __shfl_xor(rs, 8, 32);
    l_i[r] = l_i[r] * c + rs;
    corr[r] = c;
    // write P (bf16) to per-wave LDS bounce: row = M, col = key-in-block
    lPw[(r + 8 * hhi) * 32 + nlo]      = (__bf16)pa;
    lPw[(r + 8 * hhi) * 32 + 16 + nlo] = (__bf16)pb;
  }
}

__global__ __launch_bounds__(256, 1)
void fa_fwd_gqa_bf16wmma(const float* __restrict__ q,
                         const float* __restrict__ k,
                         const float* __restrict__ v,
                         float* __restrict__ out) {
  __shared__ __bf16 lK[32 * LKS];     // K tile, row-major [key][d], bf16
  __shared__ __bf16 lV[D_ * LVS];     // V tile, TRANSPOSED [d][key], bf16
  __shared__ __bf16 lP[8][16 * 32];   // per-wave P bounce buffer (C->A relayout)

  const int tid  = threadIdx.x;
  const int lane = tid & 31;
  const int wave = tid >> 5;
  const int nlo  = lane & 15;
  const int hhi  = lane >> 4;          // half-wave select (0/1)

  const int q0  = blockIdx.x * 128;    // 128 query rows per workgroup (8 waves)
  const int hq  = blockIdx.y;          // query head
  const int b   = blockIdx.z;
  const int kvh = hq / GQA_;           // GQA: repeat_interleave -> h / rep
  const int qw  = q0 + wave * 16;      // first query row of this wave

  // ---------- Q -> A-layout bf16 fragments (scale folded in) ----------
  // A 16x32 bf16 layout: lane m (and m+16) hold row m; dword j holds K pair:
  //   K(j) = (j<4 ? 2j : 2j+8) + 8*(lane>=16)
  const float* qrow = q + ((size_t)((b * S_ + qw + nlo) * H_) + hq) * D_;
  v16bf aQ[4];
#pragma unroll
  for (int f = 0; f < 4; ++f) {
#pragma unroll
    for (int j = 0; j < 8; ++j) {
      const int kk = ((j < 4) ? (2 * j) : (2 * j + 8)) + 8 * hhi;
      const float2 p = *(const float2*)(qrow + f * 32 + kk);
      aQ[f][2 * j]     = (__bf16)(p.x * SCALE_);
      aQ[f][2 * j + 1] = (__bf16)(p.y * SCALE_);
    }
  }

  // ---------- accumulators + online-softmax state ----------
  const v8f vzero = {0.f, 0.f, 0.f, 0.f, 0.f, 0.f, 0.f, 0.f};
  v8f o[8];
#pragma unroll
  for (int t = 0; t < 8; ++t) o[t] = vzero;
  float m_i[8], l_i[8];
#pragma unroll
  for (int r = 0; r < 8; ++r) { m_i[r] = -1e30f; l_i[r] = 0.f; }

  __bf16* lPw = lP[wave];

  // staging assignment: 8 threads per key row, 16-float (64 B) chunks
  const int kr = tid >> 3;             // key row this thread stages (0..31)
  const int d0 = (tid & 7) * 16;       // D chunk
  const int nkb = q0 / 32 + 4;         // causal: key blocks covering [0, q0+127]

  const size_t kvRowBase = ((size_t)((b * S_ + kr) * KVH_) + kvh) * D_ + d0;
  const size_t kvBlkStep = (size_t)32 * KVH_ * D_;   // +32 tokens

  // ---------- software pipeline: tile t+1 lives in registers during compute(t)
  float4 rK[4], rV[4];
  {
    const float* kp = k + kvRowBase;
    const float* vp = v + kvRowBase;
#pragma unroll
    for (int i = 0; i < 4; ++i) {
      rK[i] = *(const float4*)(kp + 4 * i);
      rV[i] = *(const float4*)(vp + 4 * i);
    }
  }

  for (int kb = 0; kb < nkb; ++kb) {
    const int keybase = kb * 32;

    __syncthreads();                   // previous LDS tile fully consumed

    // ---------- drain staged registers into LDS (bf16; V transposed) ----------
#pragma unroll
    for (int i = 0; i < 4; ++i) {
      const float4 x = rK[i];
      *(uint2*)(&lK[kr * LKS + d0 + 4 * i]) = make_uint2(pk2(x.x, x.y), pk2(x.z, x.w));
    }
#pragma unroll
    for (int i = 0; i < 4; ++i) {
      const float4 x = rV[i];
      const int dd = d0 + 4 * i;
      lV[(dd + 0) * LVS + kr] = (__bf16)x.x;
      lV[(dd + 1) * LVS + kr] = (__bf16)x.y;
      lV[(dd + 2) * LVS + kr] = (__bf16)x.z;
      lV[(dd + 3) * LVS + kr] = (__bf16)x.w;
    }
    __syncthreads();

    // ---------- issue global loads for tile kb+1 (latency hides under compute)
    if (kb + 1 < nkb) {
      const float* kp = k + kvRowBase + (size_t)(kb + 1) * kvBlkStep;
      const float* vp = v + kvRowBase + (size_t)(kb + 1) * kvBlkStep;
#pragma unroll
      for (int i = 0; i < 4; ++i) {
        rK[i] = *(const float4*)(kp + 4 * i);
        rV[i] = *(const float4*)(vp + 4 * i);
      }
      if (kb + 2 < nkb) {              // warm L2 one more tile ahead
        __builtin_prefetch(kp + kvBlkStep, 0, 3);
        __builtin_prefetch(vp + kvBlkStep, 0, 3);
      }
    }

    // ---------- per-wave compute (wave-uniform predicate: EXEC stays full) ----
    if (keybase <= qw + 15) {
      // S = Q * K^T : 16x32 fp32 scores via 8 bf16 WMMAs
      v8f s0 = vzero, s1 = vzero;
#pragma unroll
      for (int f = 0; f < 4; ++f) {
        Frag16 bk0, bk1;
        const __bf16* p0 = &lK[(0  + nlo) * LKS + f * 32 + 8 * hhi];
        const __bf16* p1 = &lK[(16 + nlo) * LKS + f * 32 + 8 * hhi];
        bk0.q[0] = *(const uint4*)(p0); bk0.q[1] = *(const uint4*)(p0 + 16);
        bk1.q[0] = *(const uint4*)(p1); bk1.q[1] = *(const uint4*)(p1 + 16);
        s0 = __builtin_amdgcn_wmma_f32_16x16x32_bf16(false, aQ[f], false, bk0.v,
                                                     (short)0, s0, false, false);
        s1 = __builtin_amdgcn_wmma_f32_16x16x32_bf16(false, aQ[f], false, bk1.v,
                                                     (short)0, s1, false, false);
      }

      // online softmax; masked specialization only near the diagonal
      float corr[8];
      if (keybase + 31 > qw) {
        softmax_update<true >(s0, s1, m_i, l_i, corr, lPw, nlo, hhi, qw, keybase);
      } else {
        softmax_update<false>(s0, s1, m_i, l_i, corr, lPw, nlo, hhi, qw, keybase);
      }

      // rescale running output
#pragma unroll
      for (int t = 0; t < 8; ++t)
#pragma unroll
        for (int r = 0; r < 8; ++r) o[t][r] *= corr[r];

      // read P back in A-layout (same-wave DS ops are in order)
      Frag16 aP;
      {
        const __bf16* pp = &lPw[nlo * 32 + 8 * hhi];
        aP.q[0] = *(const uint4*)(pp);
        aP.q[1] = *(const uint4*)(pp + 16);
      }

      // O += P * V : 8 bf16 WMMAs; V^T LDS layout makes B-fragments contiguous
#pragma unroll
      for (int t = 0; t < 8; ++t) {
        Frag16 bv;
        const __bf16* pv2 = &lV[(16 * t + nlo) * LVS + 8 * hhi];
        bv.q[0] = *(const uint4*)(pv2);
        bv.q[1] = *(const uint4*)(pv2 + 16);
        o[t] = __builtin_amdgcn_wmma_f32_16x16x32_bf16(false, aP.v, false, bv.v,
                                                       (short)0, o[t], false, false);
      }
    }
  }

  // ---------- normalize and store (fp32, coalesced across lanes) ----------
  float rl[8];
#pragma unroll
  for (int r = 0; r < 8; ++r) rl[r] = 1.0f / l_i[r];
#pragma unroll
  for (int t = 0; t < 8; ++t) {
#pragma unroll
    for (int r = 0; r < 8; ++r) {
      const int qr = r + 8 * hhi;
      out[((size_t)((b * S_ + qw + qr) * H_) + hq) * D_ + 16 * t + nlo] = o[t][r] * rl[r];
    }
  }
}

extern "C" void kernel_launch(void* const* d_in, const int* in_sizes, int n_in,
                              void* d_out, int out_size, void* d_ws, size_t ws_size,
                              hipStream_t stream) {
  const float* q = (const float*)d_in[0];
  const float* k = (const float*)d_in[1];
  const float* v = (const float*)d_in[2];
  float* out = (float*)d_out;
  (void)in_sizes; (void)n_in; (void)out_size; (void)d_ws; (void)ws_size;

  dim3 grid(S_ / 128, H_, B_);  // 128 query rows per workgroup
  dim3 block(256);              // 8 waves (wave32)
  fa_fwd_gqa_bf16wmma<<<grid, block, 0, stream>>>(q, k, v, out);
}
